// Block_23313082483050
// MI455X (gfx1250) — matrix operand order repaired and checked by reading, not compiled, not run on hardware
//
#include <hip/hip_runtime.h>
#include <hip/hip_bf16.h>
#include <math.h>

// Problem constants (match reference)
#define BDIM 4
#define TDIM 2048
#define SDIM 2048
#define CDIM 512
#define HHEADS 8
#define HDIM 64
#define MROWS (BDIM * TDIM)   /* 8192 == BDIM*SDIM too */
#define FDIM (4 * CDIM)       /* 2048 */

typedef __bf16 bf16_t;
typedef __attribute__((ext_vector_type(16))) __bf16 v16bf;
typedef __attribute__((ext_vector_type(8)))  __bf16 v8bf;
typedef __attribute__((ext_vector_type(8)))  float   v8f;
typedef int v4i __attribute__((ext_vector_type(4)));

#define AS1 __attribute__((address_space(1)))
#define AS3 __attribute__((address_space(3)))

#if __has_builtin(__builtin_amdgcn_global_load_async_to_lds_b128) && \
    __has_builtin(__builtin_amdgcn_s_wait_asynccnt)
#define USE_ASYNC_LDS 1
#else
#define USE_ASYNC_LDS 0
#endif

static __device__ __forceinline__ bf16_t f2bf(float f) {
  unsigned u = __builtin_bit_cast(unsigned, f);
  u += 0x7FFFu + ((u >> 16) & 1u);          // round-to-nearest-even
  unsigned short h = (unsigned short)(u >> 16);
  return __builtin_bit_cast(bf16_t, h);
}

static __device__ __forceinline__ v16bf cat8(v8bf lo, v8bf hi) {
  v16bf r;
#pragma unroll
  for (int i = 0; i < 8; ++i) { r[i] = lo[i]; r[i + 8] = hi[i]; }
  return r;
}

static __device__ __forceinline__ void wait_lds() {
#if __has_builtin(__builtin_amdgcn_s_wait_dscnt)
  __builtin_amdgcn_s_wait_dscnt(0);
#else
  asm volatile("s_wait_dscnt 0" ::: "memory");
#endif
}

// ---------------------------------------------------------------------------
// LayerNorm: fp32 in -> bf16 out, one wave32 per 512-wide row
// ---------------------------------------------------------------------------
__global__ __launch_bounds__(256)
void ln_kernel(const float* __restrict__ x, const float* __restrict__ g,
               const float* __restrict__ bta, bf16_t* __restrict__ out, int rows) {
  int wave = threadIdx.x >> 5, lane = threadIdx.x & 31;
  int row = blockIdx.x * (blockDim.x >> 5) + wave;
  if (row >= rows) return;
  const float* xr = x + (size_t)row * CDIM;
  float s = 0.f, ss = 0.f;
  for (int i = lane; i < CDIM; i += 32) { float v = xr[i]; s += v; ss += v * v; }
#pragma unroll
  for (int m = 1; m < 32; m <<= 1) {
    s += __shfl_xor(s, m, 32);
    ss += __shfl_xor(ss, m, 32);
  }
  float mu = s * (1.f / CDIM);
  float var = ss * (1.f / CDIM) - mu * mu;
  float inv = rsqrtf(var + 1e-5f);
  bf16_t* o = out + (size_t)row * CDIM;
  for (int i = lane; i < CDIM; i += 32)
    o[i] = f2bf((xr[i] - mu) * inv * g[i] + bta[i]);
}

// ---------------------------------------------------------------------------
// Weight / activation conversions to bf16
// ---------------------------------------------------------------------------
__global__ void cvt_bf16(const float* __restrict__ in, bf16_t* __restrict__ out, int n) {
  int i = blockIdx.x * 256 + threadIdx.x;
  if (i < n) out[i] = f2bf(in[i]);
}

// w: [H][C][HD] fp32 -> wt: [512][512] bf16 with wt[n][k] = w[n/HD][k][n%HD]
__global__ void cvt_wqkv(const float* __restrict__ w, bf16_t* __restrict__ wt) {
  int i = blockIdx.x * 256 + threadIdx.x;
  if (i >= CDIM * CDIM) return;
  int n = i >> 9, k = i & 511;
  wt[i] = f2bf(w[((size_t)(n >> 6) * CDIM + k) * HDIM + (n & 63)]);
}

// in [K][N] fp32 -> out [N][K] bf16
__global__ void cvt_transpose(const float* __restrict__ in, bf16_t* __restrict__ out,
                              int K, int N) {
  long i = (long)blockIdx.x * 256 + threadIdx.x;
  long total = (long)K * N;
  if (i >= total) return;
  long n = i / K, k = i % K;
  out[i] = f2bf(in[k * (long)N + n]);
}

// ---------------------------------------------------------------------------
// WMMA GEMM: C[M,N] = A[M,K] x Bt[N,K]^T (+bias)(+relu)(+residual)
// mode 0: bf16 out [M,N]   mode 1: bf16 out transposed [N,M]   mode 2: f32 out [M,N]
// 256 threads = 8 waves -> block tile 128x64; each wave a 32x32 tile = 4 WMMA/k-step
// Global->LDS via async tensor path (GLOBAL_LOAD_ASYNC_TO_LDS_B128, ASYNCcnt),
// double-buffered.
// ---------------------------------------------------------------------------
__global__ __launch_bounds__(256)
void gemm_wmma(const bf16_t* __restrict__ A, const bf16_t* __restrict__ Bt,
               const float* __restrict__ bias, const float* __restrict__ resid,
               void* __restrict__ out, int M, int N, int K, int mode, int relu) {
  __shared__ bf16_t As[2][128][40];
  __shared__ bf16_t Bs[2][64][40];
  const int tid = threadIdx.x, wave = tid >> 5, lane = tid & 31;
  const int half = lane >> 4, r = lane & 15;
  const int m0 = blockIdx.y * 128, n0 = blockIdx.x * 64;
  const int wm = (wave >> 1) * 32, wn = (wave & 1) * 32;
  // staging assignments: A 128x32 -> 2 x b128 per thread; B 64x32 -> 1 x b128
  const int arow = tid >> 1, acol = (tid & 1) * 16;
  const int brow = tid >> 2, bcol = (tid & 3) * 8;

  v8f acc[2][2];
#pragma unroll
  for (int mi = 0; mi < 2; ++mi)
#pragma unroll
    for (int ni = 0; ni < 2; ++ni) acc[mi][ni] = {};

#if USE_ASYNC_LDS
  auto issue = [&](int kk, int buf) {
    const bf16_t* ga0 = A + (size_t)(m0 + arow) * K + kk + acol;
    __builtin_amdgcn_global_load_async_to_lds_b128(
        (AS1 v4i*)ga0, (AS3 v4i*)&As[buf][arow][acol], 0, 0);
    __builtin_amdgcn_global_load_async_to_lds_b128(
        (AS1 v4i*)(ga0 + 8), (AS3 v4i*)&As[buf][arow][acol + 8], 0, 0);
    const bf16_t* gb0 = Bt + (size_t)(n0 + brow) * K + kk + bcol;
    __builtin_amdgcn_global_load_async_to_lds_b128(
        (AS1 v4i*)gb0, (AS3 v4i*)&Bs[buf][brow][bcol], 0, 0);
  };
  issue(0, 0);
#endif

  int buf = 0;
  for (int kk = 0; kk < K; kk += 32) {
#if USE_ASYNC_LDS
    if (kk + 32 < K) {
      issue(kk + 32, buf ^ 1);
      __builtin_amdgcn_s_wait_asynccnt(3);   // tile kk's 3 loads complete
    } else {
      __builtin_amdgcn_s_wait_asynccnt(0);
    }
    __syncthreads();                          // tile kk visible block-wide
#else
    *(v8bf*)&As[buf][arow][acol] =
        *(const v8bf*)(A + (size_t)(m0 + arow) * K + kk + acol);
    *(v8bf*)&As[buf][arow][acol + 8] =
        *(const v8bf*)(A + (size_t)(m0 + arow) * K + kk + acol + 8);
    *(v8bf*)&Bs[buf][brow][bcol] =
        *(const v8bf*)(Bt + (size_t)(n0 + brow) * K + kk + bcol);
    if (kk + 32 < K)
      __builtin_prefetch(A + (size_t)(m0 + arow) * K + kk + 32 + acol, 0, 3);
    __syncthreads();
#endif
    // A frags: lane(half,r) holds K = half*8+0..7 then 16+half*8+0..7
    v16bf af[2], bfm[2];
#pragma unroll
    for (int mi = 0; mi < 2; ++mi)
      af[mi] = cat8(*(const v8bf*)&As[buf][wm + mi * 16 + r][half * 8],
                    *(const v8bf*)&As[buf][wm + mi * 16 + r][16 + half * 8]);
    // B frags: lane(half,r) holds K = half*8+0..7 then 16+half*8+0..7 of column wn+ni*16+r
#pragma unroll
    for (int ni = 0; ni < 2; ++ni)
      bfm[ni] = cat8(*(const v8bf*)&Bs[buf][wn + ni * 16 + r][half * 8],
                     *(const v8bf*)&Bs[buf][wn + ni * 16 + r][16 + half * 8]);
#pragma unroll
    for (int mi = 0; mi < 2; ++mi)
#pragma unroll
      for (int ni = 0; ni < 2; ++ni)
        acc[mi][ni] = __builtin_amdgcn_wmma_f32_16x16x32_bf16(
            false, af[mi], false, bfm[ni], (short)0, acc[mi][ni], false, false);
    __syncthreads();                          // done reading buf before refill
    buf ^= 1;
  }

  // C layout: lane holds column n, rows m = base + half*8 + j
#pragma unroll
  for (int mi = 0; mi < 2; ++mi)
#pragma unroll
    for (int ni = 0; ni < 2; ++ni) {
      const int n = n0 + wn + ni * 16 + r;
      const int mbase = m0 + wm + mi * 16 + half * 8;
      const float bv = bias ? bias[n] : 0.f;
      if (mode == 2) {
        float* o = (float*)out;
#pragma unroll
        for (int j = 0; j < 8; ++j) {
          size_t idx = (size_t)(mbase + j) * N + n;
          o[idx] = acc[mi][ni][j] + bv + (resid ? resid[idx] : 0.f);
        }
      } else if (mode == 1) {
        bf16_t* o = (bf16_t*)out;
        v8bf pk;
#pragma unroll
        for (int j = 0; j < 8; ++j) pk[j] = f2bf(acc[mi][ni][j]);
        *(v8bf*)(o + (size_t)n * M + mbase) = pk;   // contiguous 16B store
      } else {
        bf16_t* o = (bf16_t*)out;
#pragma unroll
        for (int j = 0; j < 8; ++j) {
          float v = acc[mi][ni][j] + bv;
          if (relu) v = fmaxf(v, 0.f);
          o[(size_t)(mbase + j) * N + n] = f2bf(v);
        }
      }
    }
}

// ---------------------------------------------------------------------------
// Flash attention: one wave per 16-query tile per (b,h); 32-key chunks.
// Q,K: bf16 [B*rows, 512] head-major columns.  Vt: bf16 [512][B*Skv].
// O: bf16 [B*T, 512] concat-head layout.
// ---------------------------------------------------------------------------
__global__ __launch_bounds__(128)
void attn_wmma(const bf16_t* __restrict__ Q, const bf16_t* __restrict__ Km,
               const bf16_t* __restrict__ Vt, bf16_t* __restrict__ O,
               int Skv, int causal) {
  __shared__ bf16_t Pt[4][16][40];
  const int wave = threadIdx.x >> 5, lane = threadIdx.x & 31;
  const int half = lane >> 4, r = lane & 15;
  const int bh = blockIdx.y, b = bh >> 3, h = bh & 7;
  const int t0 = (blockIdx.x * 4 + wave) * 16;

  // Q fragments for both 32-wide K chunks of head dim (HD=64)
  const bf16_t* qp = Q + ((size_t)(b * TDIM + t0 + r)) * CDIM + h * HDIM;
  v16bf qa[2];
#pragma unroll
  for (int kk = 0; kk < 2; ++kk)
    qa[kk] = cat8(*(const v8bf*)(qp + kk * 32 + half * 8),
                  *(const v8bf*)(qp + kk * 32 + 16 + half * 8));

  float mi[8], li[8], al[8];
  v8f o0 = {}, o1 = {}, o2 = {}, o3 = {};
#pragma unroll
  for (int j = 0; j < 8; ++j) { mi[j] = -1e30f; li[j] = 0.f; }

  const int send = causal ? (t0 + 16) : Skv;
  const bf16_t* kb = Km + (size_t)b * Skv * CDIM + h * HDIM;
  const bf16_t* vb0 = Vt + (size_t)b * Skv;

  for (int s0 = 0; s0 < send; s0 += 32) {
    // ---- scores: Q(16x64) x K^T(64x32) as two 16x16 WMMA tiles ----
    v8f sc[2];
#pragma unroll
    for (int sub = 0; sub < 2; ++sub) {
      v8f c = {};
      const bf16_t* kp = kb + (size_t)(s0 + sub * 16 + r) * CDIM;
#pragma unroll
      for (int kk = 0; kk < 2; ++kk) {
        v16bf bbf = cat8(*(const v8bf*)(kp + kk * 32 + half * 8),
                         *(const v8bf*)(kp + kk * 32 + 16 + half * 8));
        c = __builtin_amdgcn_wmma_f32_16x16x32_bf16(false, qa[kk], false, bbf,
                                                    (short)0, c, false, false);
      }
      sc[sub] = c;
    }
    // ---- scale + causal mask ----
#pragma unroll
    for (int sub = 0; sub < 2; ++sub)
#pragma unroll
      for (int j = 0; j < 8; ++j) {
        float v = sc[sub][j] * 0.125f;  // HD^-0.5
        if (causal) {
          int col = s0 + sub * 16 + r;
          int row = t0 + half * 8 + j;
          if (col > row) v = -1e30f;
        }
        sc[sub][j] = v;
      }
    // ---- online softmax; rows live across 16 lanes (masks 1..8) ----
#pragma unroll
    for (int j = 0; j < 8; ++j) {
      float mx = fmaxf(sc[0][j], sc[1][j]);
#pragma unroll
      for (int msk = 1; msk < 16; msk <<= 1) mx = fmaxf(mx, __shfl_xor(mx, msk, 32));
      float mnew = fmaxf(mi[j], mx);
      float a = __expf(mi[j] - mnew);
      float p0 = __expf(sc[0][j] - mnew);
      float p1 = __expf(sc[1][j] - mnew);
      float rs = p0 + p1;
#pragma unroll
      for (int msk = 1; msk < 16; msk <<= 1) rs += __shfl_xor(rs, msk, 32);
      li[j] = li[j] * a + rs;
      mi[j] = mnew;
      al[j] = a;
      // stage P in C-layout -> LDS (row = half*8+j, col = sub*16 + r)
      Pt[wave][half * 8 + j][r]      = f2bf(p0);
      Pt[wave][half * 8 + j][16 + r] = f2bf(p1);
    }
#pragma unroll
    for (int j = 0; j < 8; ++j) {
      o0[j] *= al[j]; o1[j] *= al[j]; o2[j] *= al[j]; o3[j] *= al[j];
    }
    wait_lds();  // cross-lane LDS RAW within the wave (no block barrier: divergent trip counts)
    // P as A fragment (16 queries x 32 keys)
    v16bf pa = cat8(*(const v8bf*)&Pt[wave][r][half * 8],
                    *(const v8bf*)&Pt[wave][r][16 + half * 8]);
    // ---- PV: P(16x32) x V(32x64) as four 16x16 WMMA tiles; Vt rows contiguous in s ----
    const bf16_t* vp = vb0 + s0 + half * 8;
#pragma unroll
    for (int dd = 0; dd < 4; ++dd) {
      size_t vrow = (size_t)(h * HDIM + dd * 16 + r) * MROWS;
      v16bf vbf = cat8(*(const v8bf*)(vp + vrow),
                       *(const v8bf*)(vp + vrow + 16));
      v8f* oc = (dd == 0) ? &o0 : (dd == 1) ? &o1 : (dd == 2) ? &o2 : &o3;
      *oc = __builtin_amdgcn_wmma_f32_16x16x32_bf16(false, pa, false, vbf,
                                                    (short)0, *oc, false, false);
    }
  }

  // ---- normalize and store concat-head output ----
  bf16_t* op = O + ((size_t)(b * TDIM + t0 + half * 8)) * CDIM + h * HDIM + r;
#pragma unroll
  for (int j = 0; j < 8; ++j) {
    float inv = 1.f / li[j];
    op[(size_t)j * CDIM + 0]  = f2bf(o0[j] * inv);
    op[(size_t)j * CDIM + 16] = f2bf(o1[j] * inv);
    op[(size_t)j * CDIM + 32] = f2bf(o2[j] * inv);
    op[(size_t)j * CDIM + 48] = f2bf(o3[j] * inv);
  }
}

// ---------------------------------------------------------------------------
// Host orchestration
// ---------------------------------------------------------------------------
extern "C" void kernel_launch(void* const* d_in, const int* in_sizes, int n_in,
                              void* d_out, int out_size, void* d_ws, size_t ws_size,
                              hipStream_t stream) {
  const float* x     = (const float*)d_in[0];
  const float* enc   = (const float*)d_in[1];
  const float* sa_wq = (const float*)d_in[2];
  const float* sa_wk = (const float*)d_in[3];
  const float* sa_wv = (const float*)d_in[4];
  const float* sa_pw = (const float*)d_in[5];
  const float* sa_pb = (const float*)d_in[6];
  const float* ca_wq = (const float*)d_in[7];
  const float* ca_wk = (const float*)d_in[8];
  const float* ca_wv = (const float*)d_in[9];
  const float* ca_pw = (const float*)d_in[10];
  const float* ca_pb = (const float*)d_in[11];
  const float* w1    = (const float*)d_in[12];
  const float* b1    = (const float*)d_in[13];
  const float* w2    = (const float*)d_in[14];
  const float* b2    = (const float*)d_in[15];
  const float* ln1_g = (const float*)d_in[16];
  const float* ln1_b = (const float*)d_in[17];
  const float* lnc_g = (const float*)d_in[18];
  const float* lnc_b = (const float*)d_in[19];
  const float* ln2_g = (const float*)d_in[20];
  const float* ln2_b = (const float*)d_in[21];

  char* ws = (char*)d_ws;
  constexpr size_t MB = 1ull << 20;
  bf16_t* lnb  = (bf16_t*)(ws + 0 * MB);     // 8 MB  [8192,512]
  bf16_t* qb   = (bf16_t*)(ws + 8 * MB);     // 8 MB
  bf16_t* kbm  = (bf16_t*)(ws + 16 * MB);    // 8 MB
  bf16_t* vt   = (bf16_t*)(ws + 24 * MB);    // 8 MB  [512][8192]
  bf16_t* att  = (bf16_t*)(ws + 32 * MB);    // 8 MB
  bf16_t* encb = (bf16_t*)(ws + 40 * MB);    // 8 MB
  float*  xa   = (float*)(ws + 48 * MB);     // 16 MB
  float*  xb   = (float*)(ws + 64 * MB);     // 16 MB
  bf16_t* h1   = (bf16_t*)(ws + 80 * MB);    // 32 MB [8192,2048]
  bf16_t* saqT = (bf16_t*)(ws + 112 * MB);
  bf16_t* sakT = (bf16_t*)(ws + 112 * MB + 512 * 1024);
  bf16_t* savT = (bf16_t*)(ws + 113 * MB);
  bf16_t* sapT = (bf16_t*)(ws + 113 * MB + 512 * 1024);
  bf16_t* caqT = (bf16_t*)(ws + 114 * MB);
  bf16_t* cakT = (bf16_t*)(ws + 114 * MB + 512 * 1024);
  bf16_t* cavT = (bf16_t*)(ws + 115 * MB);
  bf16_t* capT = (bf16_t*)(ws + 115 * MB + 512 * 1024);
  bf16_t* w1T  = (bf16_t*)(ws + 116 * MB);   // 2 MB [2048][512]
  bf16_t* w2T  = (bf16_t*)(ws + 118 * MB);   // 2 MB [512][2048]

  {
    int g = (CDIM * CDIM + 255) / 256;
    cvt_wqkv<<<g, 256, 0, stream>>>(sa_wq, saqT);
    cvt_wqkv<<<g, 256, 0, stream>>>(sa_wk, sakT);
    cvt_wqkv<<<g, 256, 0, stream>>>(sa_wv, savT);
    cvt_wqkv<<<g, 256, 0, stream>>>(ca_wq, caqT);
    cvt_wqkv<<<g, 256, 0, stream>>>(ca_wk, cakT);
    cvt_wqkv<<<g, 256, 0, stream>>>(ca_wv, cavT);
    cvt_transpose<<<g, 256, 0, stream>>>(sa_pw, sapT, CDIM, CDIM);
    cvt_transpose<<<g, 256, 0, stream>>>(ca_pw, capT, CDIM, CDIM);
    cvt_transpose<<<(CDIM * FDIM + 255) / 256, 256, 0, stream>>>(w1, w1T, CDIM, FDIM);
    cvt_transpose<<<(FDIM * CDIM + 255) / 256, 256, 0, stream>>>(w2, w2T, FDIM, CDIM);
    cvt_bf16<<<(MROWS * CDIM + 255) / 256, 256, 0, stream>>>(enc, encb, MROWS * CDIM);
  }

  auto gemm = [&](const bf16_t* A, const bf16_t* Bt, const float* bias,
                  const float* resid, void* out, int M, int N, int K,
                  int mode, int relu) {
    dim3 grid(N / 64, M / 128);
    gemm_wmma<<<grid, 256, 0, stream>>>(A, Bt, bias, resid, out, M, N, K, mode, relu);
  };
  dim3 attnGrid(TDIM / 64, BDIM * HHEADS);

  // --- self-attention: x = x + proj(attn(ln1(x))) ---
  ln_kernel<<<MROWS / 8, 256, 0, stream>>>(x, ln1_g, ln1_b, lnb, MROWS);
  gemm(lnb, saqT, nullptr, nullptr, qb, MROWS, CDIM, CDIM, 0, 0);
  gemm(lnb, sakT, nullptr, nullptr, kbm, MROWS, CDIM, CDIM, 0, 0);
  gemm(lnb, savT, nullptr, nullptr, vt, MROWS, CDIM, CDIM, 1, 0);   // transposed V
  attn_wmma<<<attnGrid, 128, 0, stream>>>(qb, kbm, vt, att, SDIM, 1);
  gemm(att, sapT, sa_pb, x, xa, MROWS, CDIM, CDIM, 2, 0);

  // --- cross-attention: x = x + proj(attn(lnc(x), enc_out)) ---
  ln_kernel<<<MROWS / 8, 256, 0, stream>>>(xa, lnc_g, lnc_b, lnb, MROWS);
  gemm(lnb,  caqT, nullptr, nullptr, qb, MROWS, CDIM, CDIM, 0, 0);
  gemm(encb, cakT, nullptr, nullptr, kbm, MROWS, CDIM, CDIM, 0, 0);
  gemm(encb, cavT, nullptr, nullptr, vt, MROWS, CDIM, CDIM, 1, 0);
  attn_wmma<<<attnGrid, 128, 0, stream>>>(qb, kbm, vt, att, SDIM, 0);
  gemm(att, capT, ca_pb, xa, xb, MROWS, CDIM, CDIM, 2, 0);

  // --- FFN: out = x + w2(relu(w1(ln2(x)))) ---
  ln_kernel<<<MROWS / 8, 256, 0, stream>>>(xb, ln2_g, ln2_b, lnb, MROWS);
  gemm(lnb, w1T, b1, nullptr, h1, MROWS, FDIM, CDIM, 0, 1);          // bias+relu, bf16
  gemm(h1, w2T, b2, xb, d_out, MROWS, CDIM, FDIM, 2, 0);             // f32 + residual
}
